// LapEncoder_39599598469811
// MI455X (gfx1250) — matrix operand order repaired and checked by reading, not compiled, not run on hardware
//
#include <hip/hip_runtime.h>

// ---------------------------------------------------------------------------
// LapEncoder on MI455X (gfx1250, wave32, WMMA).
// - L (256MB f32) converted once to bf16 (128MB -> L2-resident at 192MB L2);
//   all 10 L@y matmuls run v_wmma_f32_16x16x32_bf16 with f32 accumulators.
// - B operands column-major bf16 => WMMA fragments are contiguous b128 loads.
// - GEMM: 4 waves/block, 64x128 output/block, K-step 64, double-buffered LDS
//   strip filled by global_load_async_to_lds_b128 (ASYNCcnt). Main loop is
//   unrolled by 2 so all double-buffer indices are compile-time constants;
//   B fragments use a 4-deep rotating register pipeline to hide LDS latency.
// - Bias/residual fused into GEMM epilogue; BN stats via atomic partials.
// ---------------------------------------------------------------------------

typedef __attribute__((ext_vector_type(16))) __bf16 v16bf;
typedef __attribute__((ext_vector_type(8)))  float  v8f;

union FragBF {
  v16bf v;
  uint4 q[2];
};

#if defined(__has_builtin)
#if __has_builtin(__builtin_amdgcn_global_load_async_to_lds_b128) && \
    __has_builtin(__builtin_amdgcn_s_wait_asynccnt)
#define ASYNC_LDS 1
#endif
#endif
#ifndef ASYNC_LDS
#define ASYNC_LDS 0
#endif

#if ASYNC_LDS
// builtin signature (from clang diagnostic): (AS1 v4i*, AS3 v4i*, imm, imm)
typedef int v4i_t __attribute__((vector_size(16)));
typedef __attribute__((address_space(1))) v4i_t* gv4p;
typedef __attribute__((address_space(3))) v4i_t* lv4p;
#endif

__device__ __forceinline__ unsigned short f2bf(float f) {
  unsigned int u = __float_as_uint(f);
  u += 0x7FFFu + ((u >> 16) & 1u);          // round-to-nearest-even
  return (unsigned short)(u >> 16);
}
__device__ __forceinline__ float eluf(float x) {
  return x > 0.f ? x : (__expf(x) - 1.f);
}

// ---- big f32 array -> bf16, 8 elements/thread, grid-stride -----------------
__global__ void k_cvt_bf16(const float* __restrict__ in,
                           unsigned short* __restrict__ out, long n8) {
  long i = (long)blockIdx.x * blockDim.x + threadIdx.x;
  long stride = (long)gridDim.x * blockDim.x;
  for (; i < n8; i += stride) {
    const float4* p = ((const float4*)in) + i * 2;
    float4 a = p[0], b = p[1];
    uint4 o;
    o.x = f2bf(a.x) | ((unsigned)f2bf(a.y) << 16);
    o.y = f2bf(a.z) | ((unsigned)f2bf(a.w) << 16);
    o.z = f2bf(b.x) | ((unsigned)f2bf(b.y) << 16);
    o.w = f2bf(b.z) | ((unsigned)f2bf(b.w) << 16);
    ((uint4*)out)[i] = o;
  }
}

// ---- weights [K][N] f32 -> transposed [N][K] bf16 (slice per blockIdx.y) ---
__global__ void k_cvt_wT(const float* __restrict__ in,
                         unsigned short* __restrict__ out, int K, int N) {
  long s = blockIdx.y;
  in  += s * (long)K * N;
  out += s * (long)K * N;
  int idx = blockIdx.x * blockDim.x + threadIdx.x;
  if (idx >= K * N) return;
  int n = idx / K, k = idx % K;
  out[idx] = f2bf(in[(long)k * N + n]);
}

// ---- conv1: [16384,3] @ [3,128] + b ---------------------------------------
__global__ void k_conv1(const float* __restrict__ inp, const float* __restrict__ W,
                        const float* __restrict__ bias, float* __restrict__ x) {
  long row = (long)blockIdx.x * 2 + (threadIdx.x >> 7);
  int c = threadIdx.x & 127;
  const float* ip = inp + row * 3;
  x[row * 128 + c] = ip[0] * W[c] + ip[1] * W[128 + c] + ip[2] * W[256 + c] + bias[c];
}

// ---- elu + transpose to column-major bf16: [b][n][128] -> [b][128][4096] ---
__global__ void k_eluT(const float* __restrict__ src, unsigned short* __restrict__ dst) {
  __shared__ float tile[32][33];
  int tx = threadIdx.x, ty = threadIdx.y;
  int n0 = blockIdx.x * 32, c0 = blockIdx.y * 32;
  long b = blockIdx.z;
  const float* X = src + b * 4096 * 128;
  unsigned short* Y = dst + b * 128 * 4096;
  for (int j = ty; j < 32; j += 8)
    tile[j][tx] = eluf(X[(long)(n0 + j) * 128 + c0 + tx]);
  __syncthreads();
  for (int j = ty; j < 32; j += 8)
    Y[(long)(c0 + j) * 4096 + n0 + tx] = f2bf(tile[tx][j]);
}

// ---- bf16 WMMA GEMM: C[M,128] = A[M,K](row-major) @ B[K,128](col-major) ----
// 128 threads = 4 waves; block covers 64 rows x 128 cols. K-step 64, main loop
// unrolled by 2 (constant buffer parity). K must be a multiple of 128.
#define BSTRIDE 72

#if ASYNC_LDS
#define STAGE_B(NXTBUF, GOFF)                                                 \
  { const unsigned short* g_ = Bcol + (GOFF);                                 \
    unsigned short* l_ = Lcol + (NXTBUF) * BUFSTRIDE;                         \
    _Pragma("unroll")                                                         \
    for (int j = 0; j < 8; ++j)                                               \
      __builtin_amdgcn_global_load_async_to_lds_b128(                         \
          (gv4p)(g_ + j * 8), (lv4p)(l_ + j * 8), 0, 0); }
#define STAGE_B_WAIT() __builtin_amdgcn_s_wait_asynccnt(0)
#else
#define STAGE_B(NXTBUF, GOFF)                                                 \
  { const unsigned short* g_ = Bcol + (GOFF);                                 \
    unsigned short* l_ = Lcol + (NXTBUF) * BUFSTRIDE;                         \
    _Pragma("unroll")                                                         \
    for (int j = 0; j < 8; ++j)                                               \
      *(uint4*)(l_ + j * 8) = *(const uint4*)(g_ + j * 8); }
#define STAGE_B_WAIT()
#endif

#define LOAD_A(DST, OFF)                                                      \
  { (DST).q[0] = *(const uint4*)(Arow + (OFF));                               \
    (DST).q[1] = *(const uint4*)(Arow + (OFF) + 8); }

// 16 WMMAs from buffer CURBUF; 4-deep rotating B-fragment pipeline (all
// indices constant after unroll -> registers, staged dscnt waits).
#define GEMM_COMPUTE(AC0, AC1, CURBUF)                                        \
  { const unsigned short* base_ = &Bs[CURBUF][0];                             \
    FragBF bf_[4];                                                            \
    _Pragma("unroll")                                                         \
    for (int pq = 0; pq < 3; ++pq) {                                          \
      const unsigned short* bp_ =                                             \
          base_ + ((pq & 7) * 16 + m) * BSTRIDE + (pq >> 3) * 32 + half * 16; \
      bf_[pq].q[0] = *(const uint4*)(bp_);                                    \
      bf_[pq].q[1] = *(const uint4*)(bp_ + 8);                                \
    }                                                                         \
    _Pragma("unroll")                                                         \
    for (int q = 0; q < 16; ++q) {                                            \
      if (q < 13) {                                                           \
        int qq = q + 3;                                                       \
        const unsigned short* bp_ =                                           \
            base_ + ((qq & 7) * 16 + m) * BSTRIDE + (qq >> 3) * 32 + half*16; \
        bf_[qq & 3].q[0] = *(const uint4*)(bp_);                              \
        bf_[qq & 3].q[1] = *(const uint4*)(bp_ + 8);                          \
      }                                                                       \
      int nt = q & 7;                                                         \
      const FragBF& aa_ = (q < 8) ? (AC0) : (AC1);                            \
      acc[nt] = __builtin_amdgcn_wmma_f32_16x16x32_bf16(                      \
          false, aa_.v, false, bf_[q & 3].v, (short)0, acc[nt], false, false);\
    } }

__global__ __launch_bounds__(128)
void k_gemm_bf16(const unsigned short* __restrict__ A,
                 const unsigned short* __restrict__ B,
                 float* __restrict__ C,
                 int K, long sA, long sB, long sC,
                 const float* __restrict__ bias,
                 const float* __restrict__ res) {
  __shared__ unsigned short Bs[2][128 * BSTRIDE];   // 36 KB
  int b = blockIdx.y;
  A += (long)b * sA; B += (long)b * sB; C += (long)b * sC;

  int lane = threadIdx.x & 31;
  int wave = threadIdx.x >> 5;     // 0..3
  int m    = lane & 15;            // A row / B col within 16x16 tile
  int half = lane >> 4;            // K-half selector (k = 16*half + s, contiguous order)

  long row = (long)blockIdx.x * 64 + wave * 16 + m;
  const unsigned short* Arow = A + row * K + half * 16;

  // B-strip loader: thread t owns column t, copies 128B (64 ushorts) per step
  int col = threadIdx.x;
  const unsigned short* Bcol = B + (long)col * K;
  unsigned short* Lcol = &Bs[0][0] + col * BSTRIDE;
  const int BUFSTRIDE = 128 * BSTRIDE;

  // ---- prologue: stage strip 0 + A fragments for step 0 ----
  STAGE_B(0, 0);
  STAGE_B_WAIT();
  FragBF a00, a01, a10, a11;
  LOAD_A(a00, 0);
  LOAD_A(a01, 32);
  __syncthreads();

  v8f acc[8] = {};
  int nsteps = K >> 6;             // even: K is a multiple of 128
  for (int s = 0; s < nsteps; s += 2) {
    int k0 = s << 6;
    // ---- even step: compute from Bs[0]/a0x, stage Bs[1]/a1x ----
    {
      STAGE_B(1, k0 + 64);
      LOAD_A(a10, k0 + 64);
      LOAD_A(a11, k0 + 96);
      __builtin_prefetch(Arow + k0 + 320, 0, 3);
      GEMM_COMPUTE(a00, a01, 0);
      STAGE_B_WAIT();
      __syncthreads();
    }
    // ---- odd step: compute from Bs[1]/a1x, stage Bs[0]/a0x ----
    {
      bool more = (s + 2 < nsteps);
      if (more) {
        STAGE_B(0, k0 + 128);
        LOAD_A(a00, k0 + 128);
        LOAD_A(a01, k0 + 160);
        __builtin_prefetch(Arow + k0 + 384, 0, 3);
      }
      GEMM_COMPUTE(a10, a11, 1);
      if (more) { STAGE_B_WAIT(); }
      __syncthreads();
    }
  }

  // epilogue: C/D layout: lanes 0-15 -> M=r, lanes 16-31 -> M=8+r; N=lane%16
  long rb = (long)blockIdx.x * 64 + wave * 16 + half * 8;
#pragma unroll
  for (int nt = 0; nt < 8; ++nt) {
    int cc = nt * 16 + m;
    float bv = bias ? bias[cc] : 0.f;
#pragma unroll
    for (int r = 0; r < 8; ++r) {
      long idx = (rb + r) * 128 + cc;
      float v = acc[nt][r] + bv;
      if (res) v += res[idx];
      C[idx] = v;
    }
  }
}

// ---- zero helper -----------------------------------------------------------
__global__ void k_zero(float* p, int n) {
  int i = blockIdx.x * blockDim.x + threadIdx.x;
  if (i < n) p[i] = 0.f;
}

// ---- BN stats over 16384 rows: 256 channels (elu(X) | LY) ------------------
__global__ void k_bn_stats(const float* __restrict__ X, const float* __restrict__ LY,
                           float* __restrict__ sums) {
  int c = threadIdx.x;                 // 0..255
  long r0 = (long)blockIdx.x * 64;
  float s = 0.f, s2 = 0.f;
  if (c < 128) {
    for (int r = 0; r < 64; ++r) { float v = eluf(X[(r0 + r) * 128 + c]); s += v; s2 += v * v; }
  } else {
    int cc = c - 128;
    for (int r = 0; r < 64; ++r) { float v = LY[(r0 + r) * 128 + cc]; s += v; s2 += v * v; }
  }
  atomicAdd(&sums[c], s);
  atomicAdd(&sums[256 + c], s2);
}

// ---- BN stats, 128 channels (elu(X) only) ----------------------------------
__global__ void k_bn_stats128(const float* __restrict__ X, float* __restrict__ sums) {
  int c = threadIdx.x & 127;
  long r0 = (long)blockIdx.x * 64 + (threadIdx.x >> 7) * 32;
  float s = 0.f, s2 = 0.f;
  for (int r = 0; r < 32; ++r) { float v = eluf(X[(r0 + r) * 128 + c]); s += v; s2 += v * v; }
  atomicAdd(&sums[c], s);
  atomicAdd(&sums[256 + c], s2);
}

// ---- normalize concat[elu(X), LY] -> bf16 [16384,256] ----------------------
__global__ void k_bn_norm(const float* __restrict__ X, const float* __restrict__ LY,
                          const float* __restrict__ sums,
                          const float* __restrict__ gamma, const float* __restrict__ beta,
                          unsigned short* __restrict__ out) {
  int c = threadIdx.x;
  long row = blockIdx.x;
  float mean = sums[c] * (1.f / 16384.f);
  float var  = sums[256 + c] * (1.f / 16384.f) - mean * mean;
  float rs   = rsqrtf(var + 1e-5f);
  float v = (c < 128) ? eluf(X[row * 128 + c]) : LY[row * 128 + (c - 128)];
  out[row * 256 + c] = f2bf(gamma[c] * (v - mean) * rs + beta[c]);
}

// ---- normalize elu(X) -> bf16 [16384,128] ----------------------------------
__global__ void k_bn_norm128(const float* __restrict__ X, const float* __restrict__ sums,
                             const float* __restrict__ gamma, const float* __restrict__ beta,
                             unsigned short* __restrict__ out) {
  int c = threadIdx.x & 127;
  long row = (long)blockIdx.x * 2 + (threadIdx.x >> 7);
  float mean = sums[c] * (1.f / 16384.f);
  float var  = sums[256 + c] * (1.f / 16384.f) - mean * mean;
  float rs   = rsqrtf(var + 1e-5f);
  float v = eluf(X[row * 128 + c]);
  out[row * 128 + c] = f2bf(gamma[c] * (v - mean) * rs + beta[c]);
}

// ---- masked pool partials: elu(g)*mask summed over a 256-row slab ----------
__global__ void k_pool_partial(const float* __restrict__ g, const float* __restrict__ mask,
                               float* __restrict__ praw, float* __restrict__ msum) {
  int b = blockIdx.y, c = threadIdx.x;   // 128 threads
  long n0 = (long)blockIdx.x * 256;
  float acc = 0.f, ms = 0.f;
  for (int n = 0; n < 256; ++n) {
    long rowg = (long)b * 4096 + n0 + n;
    float mk = mask[rowg];
    float v  = eluf(g[rowg * 128 + c]);
    acc += v * mk;
    ms  += mk;
  }
  atomicAdd(&praw[b * 128 + c], acc);
  if (c == 0) atomicAdd(&msum[b], ms);
}

// ---- mu / logvar heads: (praw/msum) @ [128,100] (+b) x2 --------------------
__global__ void k_heads(const float* __restrict__ praw, const float* __restrict__ msum,
                        const float* __restrict__ muW, const float* __restrict__ mub,
                        const float* __restrict__ lvW, const float* __restrict__ lvb,
                        float* __restrict__ out) {
  int t = blockIdx.x * blockDim.x + threadIdx.x;
  if (t >= 800) return;
  int b = t / 200, j = t % 200;
  const float *W, *bb;
  int o;
  if (j < 100) { W = muW; bb = mub; o = b * 100 + j; }
  else         { j -= 100; W = lvW; bb = lvb; o = 400 + b * 100 + j; }
  float dot = 0.f;
  for (int k = 0; k < 128; ++k) dot += praw[b * 128 + k] * W[k * 100 + j];
  out[o] = bb[j] + dot / msum[b];
}

// ---------------------------------------------------------------------------
extern "C" void kernel_launch(void* const* d_in, const int* in_sizes, int n_in,
                              void* d_out, int out_size, void* d_ws, size_t ws_size,
                              hipStream_t stream) {
  const float* inputs   = (const float*)d_in[0];
  const float* L        = (const float*)d_in[1];
  const float* mask     = (const float*)d_in[2];
  const float* conv1_W  = (const float*)d_in[3];
  const float* conv1_b  = (const float*)d_in[4];
  const float* rn_gamma = (const float*)d_in[5];
  const float* rn_beta  = (const float*)d_in[6];
  const float* rn_W     = (const float*)d_in[7];
  const float* rn_b     = (const float*)d_in[8];
  const float* bn2_g    = (const float*)d_in[9];
  const float* bn2_b    = (const float*)d_in[10];
  const float* conv2_W  = (const float*)d_in[11];
  const float* conv2_b  = (const float*)d_in[12];
  const float* mu_W     = (const float*)d_in[13];
  const float* mu_b     = (const float*)d_in[14];
  const float* lv_W     = (const float*)d_in[15];
  const float* lv_b     = (const float*)d_in[16];
  float* out = (float*)d_out;
  (void)in_sizes; (void)n_in; (void)out_size; (void)ws_size;

  // workspace carve (256B aligned)
  char* p = (char*)d_ws;
  auto carve = [&](size_t bytes) { char* r = p; p += (bytes + 255) & ~(size_t)255; return r; };
  unsigned short* Lbf  = (unsigned short*)carve(4ULL * 4096 * 4096 * 2);  // 128 MB, L2-resident
  float*          xA   = (float*)carve(16384ULL * 128 * 4);
  float*          t1   = (float*)carve(16384ULL * 128 * 4);
  unsigned short* ybf  = (unsigned short*)carve(4ULL * 128 * 4096 * 2);   // y^T, col-major
  float*          Lyf  = (float*)carve(16384ULL * 128 * 4);
  unsigned short* ncat = (unsigned short*)carve(16384ULL * 256 * 2);
  unsigned short* nbf  = (unsigned short*)carve(16384ULL * 128 * 2);
  float*          gout = (float*)carve(16384ULL * 128 * 4);
  unsigned short* WtRn = (unsigned short*)carve(10ULL * 256 * 128 * 2);   // transposed bf16
  unsigned short* WtC2 = (unsigned short*)carve(128ULL * 128 * 2);
  float*          sums = (float*)carve(512 * 4);
  float*          pz   = (float*)carve(520 * 4);   // praw[512] + msum[4]
  float*          praw = pz;
  float*          msum = pz + 512;

  // one-time conversions
  k_cvt_bf16<<<4096, 256, 0, stream>>>(L, Lbf, 4LL * 4096 * 4096 / 8);
  k_cvt_wT<<<dim3(128, 10), 256, 0, stream>>>(rn_W, WtRn, 256, 128);
  k_cvt_wT<<<dim3(64, 1), 256, 0, stream>>>(conv2_W, WtC2, 128, 128);
  k_conv1<<<8192, 256, 0, stream>>>(inputs, conv1_W, conv1_b, xA);

  for (int i = 0; i < 5; ++i) {
    for (int h = 0; h < 2; ++h) {
      int s = i * 2 + h;
      const float* src = (h == 0) ? xA : t1;
      k_eluT<<<dim3(128, 4, 4), dim3(32, 8), 0, stream>>>(src, ybf);
      // Ly = L @ elu(src), bf16 WMMA, per-batch
      k_gemm_bf16<<<dim3(64, 4), 128, 0, stream>>>(
          Lbf, ybf, Lyf, 4096, 4096LL * 4096, 128LL * 4096, 4096LL * 128,
          nullptr, nullptr);
      k_zero<<<2, 256, 0, stream>>>(sums, 512);
      k_bn_stats<<<256, 256, 0, stream>>>(src, Lyf, sums);
      k_bn_norm<<<16384, 256, 0, stream>>>(src, Lyf, sums,
                                           rn_gamma + s * 256, rn_beta + s * 256, ncat);
      // [16384,256] @ [256,128] bf16 WMMA, bias (and residual) fused
      if (h == 0)
        k_gemm_bf16<<<dim3(256, 1), 128, 0, stream>>>(
            ncat, WtRn + (long)s * 256 * 128, t1, 256, 0, 0, 0,
            rn_b + s * 128, nullptr);
      else
        k_gemm_bf16<<<dim3(256, 1), 128, 0, stream>>>(
            ncat, WtRn + (long)s * 256 * 128, xA, 256, 0, 0, 0,
            rn_b + s * 128, xA);
    }
  }

  // final: bn2 -> conv2(+bias fused) -> elu -> masked pool -> heads
  k_zero<<<2, 256, 0, stream>>>(sums, 512);
  k_bn_stats128<<<256, 256, 0, stream>>>(xA, sums);
  k_bn_norm128<<<8192, 256, 0, stream>>>(xA, sums, bn2_g, bn2_b, nbf);
  k_gemm_bf16<<<dim3(256, 1), 128, 0, stream>>>(
      nbf, WtC2, gout, 128, 0, 0, 0, conv2_b, nullptr);
  k_zero<<<3, 256, 0, stream>>>(pz, 520);
  k_pool_partial<<<dim3(16, 4), 128, 0, stream>>>(gout, mask, praw, msum);
  k_heads<<<4, 256, 0, stream>>>(praw, msum, mu_W, mu_b, lv_W, lv_b, out);
}
